// Model_31817117729576
// MI455X (gfx1250) — compile-verified, matrix operand order checked by reading, not run on hardware
//
#include <hip/hip_runtime.h>
#include <hip/hip_bf16.h>

// ---------------- model dimensions ----------------
#define NCn   20000
#define NTn   128
#define ECn   320000
#define ETn   2048
#define GCn   2000
#define GTn   64
#define HD    256
#define EHD   8
#define QKD   64
#define OUTD  1024
#define KGP   288   // gconv K: 256+8=264 padded to multiple of 32

typedef __attribute__((ext_vector_type(16))) __bf16 bf16x16;
typedef __attribute__((ext_vector_type(8)))  float  v8f;
struct alignas(16) B128 { unsigned int w[4]; };

static inline int cdiv(int a, int b) { return (a + b - 1) / b; }

__device__ __forceinline__ void load_frag(bf16x16& f, const __hip_bfloat16* p) {
    B128 lo = *reinterpret_cast<const B128*>(p);
    B128 hi = *reinterpret_cast<const B128*>(p + 16);
    __builtin_memcpy(&f, &lo, 16);
    __builtin_memcpy(reinterpret_cast<char*>(&f) + 16, &hi, 16);
}

#define WMMA_BF16(a, b, c) \
    __builtin_amdgcn_wmma_f32_16x16x32_bf16(false, (a), false, (b), (short)0, (c), false, false)

struct Frags { bf16x16 a0, a1, b0, b1, b2, b3; };

// ---------------- generic WMMA GEMM ----------------
// C[M,N] = act( A[M,K](bf16,row-major) * Bt[N,K]^T (bf16,row-major) + bias + resid )
// Wave tile 32x64 (2 M-subtiles x 4 N-subtiles), block tile 128x64 (4 waves).
// Wave index is forced into an SGPR via readfirstlane so all tile guards are
// scalar branches: EXEC stays all-ones around every WMMA (ISA requirement).
// Fast path (interior tiles): branch-free K-loop with double-buffered fragments
// so global_load_b128 of slab k+1 overlaps the 8 WMMAs of slab k.
// Optional epilogues: f32 out, bf16 out, bf16 transposed out, placement mask,
// atomic row-scatter (segment_sum fusion). M,N multiples of 16; K multiple of 32.
__global__ __launch_bounds__(128)
void k_gemm(const __hip_bfloat16* __restrict__ A, int lda,
            const __hip_bfloat16* __restrict__ Bt, int ldb,
            const float* __restrict__ bias,
            int M, int N, int K,
            float* __restrict__ outF, int ldoF,
            const float* __restrict__ residual, int ldr,
            __hip_bfloat16* __restrict__ outB, int ldoB,
            __hip_bfloat16* __restrict__ outBT, int ldoBT,
            const unsigned char* __restrict__ mask, int maskMode, int maskLd,
            const int* __restrict__ srows, float* __restrict__ sout, int ldsc,
            int actTanh)
{
    // wave id is uniform per wave: pin it to an SGPR so tile guards are scalar.
    const int wave = __builtin_amdgcn_readfirstlane((int)(threadIdx.x >> 5));
    const int lane = threadIdx.x & 31;
    const int half = lane >> 4;
    const int l16  = lane & 15;
    const int rowBase = blockIdx.y * 128 + wave * 32;
    if (rowBase >= M) return;                       // scalar branch
    const bool row1ok = (rowBase + 16) < M;
    const int colBase = blockIdx.x * 64;
    int nTiles = (N - colBase) >> 4;                // exact: N multiple of 16
    if (nTiles > 4) nTiles = 4;

    v8f acc[2][4];
#pragma unroll
    for (int mi = 0; mi < 2; ++mi)
#pragma unroll
        for (int j = 0; j < 4; ++j)
#pragma unroll
            for (int v = 0; v < 8; ++v) acc[mi][j][v] = 0.0f;

    // Fragment base pointers.  A: lane l16 holds row (rowBase+l16); half-wave
    // k-chunks at 8*half and 8*half+16 within each 32-wide K slab (ISA layout).
    const __hip_bfloat16* aRow0 = A + (size_t)(rowBase + l16) * lda + 8 * half;
    const __hip_bfloat16* aRow1 = aRow0 + (size_t)16 * lda;
    const __hip_bfloat16* bRow0 = Bt + (size_t)(colBase + 0  + l16) * ldb + 8 * half;
    const __hip_bfloat16* bRow1 = Bt + (size_t)(colBase + 16 + l16) * ldb + 8 * half;
    const __hip_bfloat16* bRow2 = Bt + (size_t)(colBase + 32 + l16) * ldb + 8 * half;
    const __hip_bfloat16* bRow3 = Bt + (size_t)(colBase + 48 + l16) * ldb + 8 * half;

    if (row1ok && nTiles == 4) {
        // ---- fast path: full 32x64 tile, double-buffered K pipeline ----
        Frags cur, nxt;
        load_frag(cur.a0, aRow0); load_frag(cur.a1, aRow1);
        load_frag(cur.b0, bRow0); load_frag(cur.b1, bRow1);
        load_frag(cur.b2, bRow2); load_frag(cur.b3, bRow3);
        int kc = 0;
        while (true) {
            int kn = kc + 32;
            bool more = kn < K;
            if (more) {
                load_frag(nxt.a0, aRow0 + kn); load_frag(nxt.a1, aRow1 + kn);
                load_frag(nxt.b0, bRow0 + kn); load_frag(nxt.b1, bRow1 + kn);
                load_frag(nxt.b2, bRow2 + kn); load_frag(nxt.b3, bRow3 + kn);
            }
            acc[0][0] = WMMA_BF16(cur.a0, cur.b0, acc[0][0]);
            acc[0][1] = WMMA_BF16(cur.a0, cur.b1, acc[0][1]);
            acc[0][2] = WMMA_BF16(cur.a0, cur.b2, acc[0][2]);
            acc[0][3] = WMMA_BF16(cur.a0, cur.b3, acc[0][3]);
            acc[1][0] = WMMA_BF16(cur.a1, cur.b0, acc[1][0]);
            acc[1][1] = WMMA_BF16(cur.a1, cur.b1, acc[1][1]);
            acc[1][2] = WMMA_BF16(cur.a1, cur.b2, acc[1][2]);
            acc[1][3] = WMMA_BF16(cur.a1, cur.b3, acc[1][3]);
            if (!more) break;
            int kn2 = kn + 32;
            bool more2 = kn2 < K;
            if (more2) {
                load_frag(cur.a0, aRow0 + kn2); load_frag(cur.a1, aRow1 + kn2);
                load_frag(cur.b0, bRow0 + kn2); load_frag(cur.b1, bRow1 + kn2);
                load_frag(cur.b2, bRow2 + kn2); load_frag(cur.b3, bRow3 + kn2);
            }
            acc[0][0] = WMMA_BF16(nxt.a0, nxt.b0, acc[0][0]);
            acc[0][1] = WMMA_BF16(nxt.a0, nxt.b1, acc[0][1]);
            acc[0][2] = WMMA_BF16(nxt.a0, nxt.b2, acc[0][2]);
            acc[0][3] = WMMA_BF16(nxt.a0, nxt.b3, acc[0][3]);
            acc[1][0] = WMMA_BF16(nxt.a1, nxt.b0, acc[1][0]);
            acc[1][1] = WMMA_BF16(nxt.a1, nxt.b1, acc[1][1]);
            acc[1][2] = WMMA_BF16(nxt.a1, nxt.b2, acc[1][2]);
            acc[1][3] = WMMA_BF16(nxt.a1, nxt.b3, acc[1][3]);
            if (!more2) break;
            kc = kn2;
        }
    } else {
        // ---- edge path: scalar-guarded, single-buffered ----
        const __hip_bfloat16* bRows[4] = { bRow0, bRow1, bRow2, bRow3 };
        for (int kc = 0; kc < K; kc += 32) {
            bf16x16 a0, a1;
            load_frag(a0, aRow0 + kc);
            if (row1ok) load_frag(a1, aRow1 + kc);
#pragma unroll
            for (int j = 0; j < 4; ++j) {
                if (j >= nTiles) break;             // scalar branch
                bf16x16 bf;
                load_frag(bf, bRows[j] + kc);
                acc[0][j] = WMMA_BF16(a0, bf, acc[0][j]);
                if (row1ok) acc[1][j] = WMMA_BF16(a1, bf, acc[1][j]);
            }
        }
    }

    // Epilogue.  D layout: col = colBase+j*16+l16 ; row = base + v + 8*half.
#pragma unroll
    for (int mi = 0; mi < 2; ++mi) {
        const int rb = rowBase + mi * 16;
        if (rb >= M) break;                          // scalar branch
#pragma unroll
        for (int j = 0; j < 4; ++j) {
            if (j >= nTiles) break;
            const int col = colBase + j * 16 + l16;
#pragma unroll
            for (int v = 0; v < 8; ++v) {
                const int row = rb + v + 8 * half;
                float val = acc[mi][j][v];
                if (bias) val += bias[col];
                if (maskMode == 1) {                 // mask[row, col]
                    if (!mask[(size_t)row * maskLd + col]) val -= 1e9f;
                } else if (maskMode == 2) {          // mask[col, row] (transposed)
                    if (!mask[(size_t)col * maskLd + row]) val -= 1e9f;
                }
                if (residual) val += residual[(size_t)row * ldr + col];
                if (actTanh) val = tanhf(val);
                if (outF)  outF[(size_t)row * ldoF + col] = val;
                if (outB)  outB[(size_t)row * ldoB + col] = __float2bfloat16(val);
                if (outBT) outBT[(size_t)col * ldoBT + row] = __float2bfloat16(val);
                if (sout)  atomicAdd(&sout[(size_t)srows[row] * ldsc + col], val);
            }
        }
    }
}

// ---------------- small helper kernels ----------------
__global__ void k_zero(float* __restrict__ p, long long n) {
    long long i = (long long)blockIdx.x * blockDim.x + threadIdx.x;
    if (i < n) p[i] = 0.0f;
}

__global__ void k_f2b(const float* __restrict__ in, __hip_bfloat16* __restrict__ out, long long n) {
    long long i = (long long)blockIdx.x * blockDim.x + threadIdx.x;
    if (i < n) out[i] = __float2bfloat16(in[i]);
}

// weight f32 [K0,N] -> bf16 transposed [N,Kp] (zero-padded K)
__global__ void k_cvt_w(const float* __restrict__ W, __hip_bfloat16* __restrict__ Wt,
                        int K0, int N, int Kp) {
    int id = blockIdx.x * blockDim.x + threadIdx.x;
    if (id >= N * Kp) return;
    int n = id / Kp, k = id % Kp;
    float v = (k < K0) ? W[(size_t)k * N + n] : 0.0f;
    Wt[id] = __float2bfloat16(v);
}

// A_c bf16 [NC,32] = concat(cfeats[NC,24], op_emb[ctypes][NC,8])
__global__ void k_build_Ac(const float* __restrict__ cfeats, const float* __restrict__ op_emb,
                           const int* __restrict__ ctypes, __hip_bfloat16* __restrict__ A, int M) {
    int id = blockIdx.x * blockDim.x + threadIdx.x;
    if (id >= M * 32) return;
    int m = id >> 5, c = id & 31;
    float v = (c < 24) ? cfeats[(size_t)m * 24 + c]
                       : op_emb[(size_t)ctypes[m] * 8 + (c - 24)];
    A[id] = __float2bfloat16(v);
}

// A_t bf16 [NT,32] = tfeats[NT,16] zero-padded
__global__ void k_build_At(const float* __restrict__ tfeats, __hip_bfloat16* __restrict__ A, int M) {
    int id = blockIdx.x * blockDim.x + threadIdx.x;
    if (id >= M * 32) return;
    int m = id >> 5, c = id & 31;
    float v = (c < 16) ? tfeats[(size_t)m * 16 + c] : 0.0f;
    A[id] = __float2bfloat16(v);
}

// edge embed: out[e][j] = tanh(sum_k ef[e,k]*W[k,j] + b[j]),  4 -> 8
__global__ void k_edge_embed(const float* __restrict__ ef, const float* __restrict__ W,
                             const float* __restrict__ b, float* __restrict__ out, int E) {
    int id = blockIdx.x * blockDim.x + threadIdx.x;
    if (id >= E * 8) return;
    int e = id >> 3, j = id & 7;
    float acc = b[j];
#pragma unroll
    for (int k = 0; k < 4; ++k) acc += ef[(size_t)e * 4 + k] * W[k * 8 + j];
    out[id] = tanhf(acc);
}

// per-edge gather/scatter-sum:  agg[dst] += concat(nodeF[src], edgeF[e])
__global__ void k_scatter_edges(const float* __restrict__ nodeF, int ldn,
                                const float* __restrict__ edgeF, int eh,
                                const int* __restrict__ src, const int* __restrict__ dst,
                                float* __restrict__ agg, int ldagg, int E, int Hn) {
    int e = blockIdx.x;
    if (e >= E) return;
    int s = src[e], d = dst[e];
    const float* nf = nodeF + (size_t)s * ldn;
    float* ag = agg + (size_t)d * ldagg;
    for (int t = threadIdx.x; t < Hn; t += blockDim.x) atomicAdd(&ag[t], nf[t]);
    for (int t = threadIdx.x; t < eh; t += blockDim.x)
        atomicAdd(&ag[Hn + t], edgeF[(size_t)e * eh + t]);
}

// row-wise softmax: P(bf16) = softmax(S(f32)) per row
__global__ __launch_bounds__(256)
void k_softmax_rows(const float* __restrict__ S, __hip_bfloat16* __restrict__ P, int N) {
    __shared__ float red[256];
    const int row = blockIdx.x;
    const float* s = S + (size_t)row * N;
    __hip_bfloat16* p = P + (size_t)row * N;
    float m = -3.0e38f;
    for (int i = threadIdx.x; i < N; i += 256) m = fmaxf(m, s[i]);
    red[threadIdx.x] = m; __syncthreads();
    for (int st = 128; st > 0; st >>= 1) {
        if (threadIdx.x < st) red[threadIdx.x] = fmaxf(red[threadIdx.x], red[threadIdx.x + st]);
        __syncthreads();
    }
    m = red[0]; __syncthreads();
    float sum = 0.0f;
    for (int i = threadIdx.x; i < N; i += 256) sum += __expf(s[i] - m);
    red[threadIdx.x] = sum; __syncthreads();
    for (int st = 128; st > 0; st >>= 1) {
        if (threadIdx.x < st) red[threadIdx.x] += red[threadIdx.x + st];
        __syncthreads();
    }
    const float inv = 1.0f / red[0];
    for (int i = threadIdx.x; i < N; i += 256)
        p[i] = __float2bfloat16(__expf(s[i] - m) * inv);
}

// ---------------- host-side launch helpers ----------------
static void launch_gemm(hipStream_t st,
    const __hip_bfloat16* A, int lda, const __hip_bfloat16* Bt, int ldb,
    const float* bias, int M, int N, int K,
    float* outF, int ldoF, const float* res, int ldr,
    __hip_bfloat16* outB, int ldoB, __hip_bfloat16* outBT, int ldoBT,
    const unsigned char* mask, int maskMode, int maskLd,
    const int* srows, float* sout, int ldsc, int act)
{
    dim3 grid(cdiv(N, 64), cdiv(M, 128));
    k_gemm<<<grid, 128, 0, st>>>(A, lda, Bt, ldb, bias, M, N, K, outF, ldoF, res, ldr,
                                 outB, ldoB, outBT, ldoBT, mask, maskMode, maskLd,
                                 srows, sout, ldsc, act);
}

static void cvtw(hipStream_t st, const float* W, __hip_bfloat16* Wt, int K0, int N, int Kp) {
    int n = N * Kp;
    k_cvt_w<<<cdiv(n, 256), 256, 0, st>>>(W, Wt, K0, N, Kp);
}

static void zero_f(hipStream_t st, float* p, long long n) {
    k_zero<<<(int)((n + 255) / 256), 256, 0, st>>>(p, n);
}
static void f2b(hipStream_t st, const float* in, __hip_bfloat16* out, long long n) {
    k_f2b<<<(int)((n + 255) / 256), 256, 0, st>>>(in, out, n);
}

// one cross-attention block (both branches + output projection)
static void do_cross(hipStream_t st, float* thisF, __hip_bfloat16* thisB, int M,
                     const __hip_bfloat16* thatB, int Nk,
                     __hip_bfloat16* const* wqT, const float* const* bq,
                     __hip_bfloat16* const* wkT, const float* const* bk,
                     __hip_bfloat16* const* wvT, const float* const* bv,
                     const __hip_bfloat16* wdT, const float* bd,
                     const unsigned char* mask, int maskMode,
                     __hip_bfloat16* Q, __hip_bfloat16* Kb, __hip_bfloat16* Vt,
                     float* S, __hip_bfloat16* P, __hip_bfloat16* cat)
{
    for (int s = 0; s < 2; ++s) {
        // K_s = that @ Wk_s + bk_s    -> bf16 [Nk,64]
        launch_gemm(st, thatB, HD, wkT[s], HD, bk[s], Nk, QKD, HD,
                    nullptr, 0, nullptr, 0, Kb, QKD, nullptr, 0,
                    nullptr, 0, 0, nullptr, nullptr, 0, 0);
        // V_s^T = (that @ Wv_s + bv_s)^T -> bf16 [256, Nk]
        launch_gemm(st, thatB, HD, wvT[s], HD, bv[s], Nk, HD, HD,
                    nullptr, 0, nullptr, 0, nullptr, 0, Vt, Nk,
                    nullptr, 0, 0, nullptr, nullptr, 0, 0);
        // Q_s = this @ Wq_s + bq_s    -> bf16 [M,64]
        launch_gemm(st, thisB, HD, wqT[s], HD, bq[s], M, QKD, HD,
                    nullptr, 0, nullptr, 0, Q, QKD, nullptr, 0,
                    nullptr, 0, 0, nullptr, nullptr, 0, 0);
        // S = Q @ K^T (branch 2 masked)  -> f32 [M,Nk]
        launch_gemm(st, Q, QKD, Kb, QKD, nullptr, M, Nk, QKD,
                    S, Nk, nullptr, 0, nullptr, 0, nullptr, 0,
                    (s == 1) ? mask : nullptr, (s == 1) ? maskMode : 0, NTn,
                    nullptr, nullptr, 0, 0);
        // P = softmax(S) -> bf16 [M,Nk]
        k_softmax_rows<<<M, 256, 0, st>>>(S, P, Nk);
        // R_s = P @ V -> bf16 into cat[:, s*256 : s*256+256]
        launch_gemm(st, P, Nk, Vt, Nk, nullptr, M, HD, Nk,
                    nullptr, 0, nullptr, 0, cat + s * HD, 2 * HD, nullptr, 0,
                    nullptr, 0, 0, nullptr, nullptr, 0, 0);
    }
    // this = tanh(cat @ Wd + bd + this)  (in place; refresh bf16 copy)
    launch_gemm(st, cat, 2 * HD, wdT, 2 * HD, bd, M, HD, 2 * HD,
                thisF, HD, thisF, HD, thisB, HD, nullptr, 0,
                nullptr, 0, 0, nullptr, nullptr, 0, 1);
}

// ---------------- input index map (setup_inputs order, params in insertion order) ----
enum {
    IN_CFEATS = 0, IN_CEDGE, IN_TFEATS, IN_TEDGE, IN_PLACEMENT,
    P_OP_EMB, P_CN_W, P_CN_B, P_CE_W, P_CE_B, P_TN_W, P_TN_B, P_TE_W, P_TE_B,
    P_CG_W, P_CG_B, P_TG_W, P_TG_B, P_CF_W, P_CF_B, P_TF_W, P_TF_B,
    P_CC_BASE = 22,   // Wq1,bq1,Wk1,bk1,Wv1,bv1,Wq2,bq2,Wk2,bk2,Wv2,bv2,Wd,bd
    P_TC_BASE = 36,
    IN_CTYPES = 50, IN_CSRC, IN_CDST, IN_TSRC, IN_TDST, IN_CGID, IN_TGID,
    IN_COUNT = 57
};

extern "C" void kernel_launch(void* const* d_in, const int* in_sizes, int n_in,
                              void* d_out, int out_size, void* d_ws, size_t ws_size,
                              hipStream_t stream)
{
    (void)in_sizes; (void)out_size; (void)ws_size;
    if (n_in < IN_COUNT) return;

    auto F = [&](int i) { return (const float*)d_in[i]; };
    const unsigned char* placement = (const unsigned char*)d_in[IN_PLACEMENT];
    const int* ctypes = (const int*)d_in[IN_CTYPES];
    const int* c_src = (const int*)d_in[IN_CSRC];
    const int* c_dst = (const int*)d_in[IN_CDST];
    const int* t_src = (const int*)d_in[IN_TSRC];
    const int* t_dst = (const int*)d_in[IN_TDST];
    const int* c_gid = (const int*)d_in[IN_CGID];
    const int* t_gid = (const int*)d_in[IN_TGID];

    // ---- workspace carve-out (aligned 256B) ----
    char* base = (char*)d_ws;
    size_t off = 0;
    auto alloc = [&](size_t bytes) -> void* {
        off = (off + 255) & ~(size_t)255;
        void* p = base + off; off += bytes; return p;
    };
    auto bfbuf = [&](size_t e) { return (__hip_bfloat16*)alloc(e * sizeof(__hip_bfloat16)); };
    auto f32buf = [&](size_t e) { return (float*)alloc(e * sizeof(float)); };

    // weights (bf16, transposed [N,Kp])
    __hip_bfloat16* cnWt = bfbuf((size_t)HD * 32);
    __hip_bfloat16* tnWt = bfbuf((size_t)HD * 32);
    __hip_bfloat16* cgWt[2] = { bfbuf((size_t)HD * KGP), bfbuf((size_t)HD * KGP) };
    __hip_bfloat16* tgWt[2] = { bfbuf((size_t)HD * KGP), bfbuf((size_t)HD * KGP) };
    __hip_bfloat16* cfWt = bfbuf((size_t)OUTD * HD);
    __hip_bfloat16* tfWt = bfbuf((size_t)OUTD * HD);
    __hip_bfloat16 *wq1t[2][2], *wk1t[2][2], *wv1t[2][2];
    __hip_bfloat16 *wq2t[2][2], *wk2t[2][2], *wv2t[2][2], *wdt[2][2];
    for (int p = 0; p < 2; ++p)
        for (int l = 0; l < 2; ++l) {
            wq1t[p][l] = bfbuf((size_t)QKD * HD);
            wk1t[p][l] = bfbuf((size_t)QKD * HD);
            wv1t[p][l] = bfbuf((size_t)HD * HD);
            wq2t[p][l] = bfbuf((size_t)QKD * HD);
            wk2t[p][l] = bfbuf((size_t)QKD * HD);
            wv2t[p][l] = bfbuf((size_t)HD * HD);
            wdt[p][l]  = bfbuf((size_t)HD * 2 * HD);
        }

    // activations & scratch
    __hip_bfloat16* Ac = bfbuf((size_t)NCn * 32);
    __hip_bfloat16* At = bfbuf((size_t)NTn * 32);
    float* cF = f32buf((size_t)NCn * HD);   __hip_bfloat16* cB = bfbuf((size_t)NCn * HD);
    float* tF = f32buf((size_t)NTn * HD);   __hip_bfloat16* tB = bfbuf((size_t)NTn * HD);
    float* ceF = f32buf((size_t)ECn * EHD);
    float* teF = f32buf((size_t)ETn * EHD);
    float* aggC = f32buf((size_t)NCn * KGP); __hip_bfloat16* aggCB = bfbuf((size_t)NCn * KGP);
    float* aggT = f32buf((size_t)NTn * KGP); __hip_bfloat16* aggTB = bfbuf((size_t)NTn * KGP);
    __hip_bfloat16* Qs = bfbuf((size_t)NCn * QKD);
    __hip_bfloat16* Ks = bfbuf((size_t)NCn * QKD);
    __hip_bfloat16* Vt = bfbuf((size_t)HD * NCn);
    float* S = f32buf((size_t)NCn * NTn);
    __hip_bfloat16* P = bfbuf((size_t)NCn * NTn);
    __hip_bfloat16* cat = bfbuf((size_t)NCn * 2 * HD);
    float* cg = f32buf((size_t)GCn * OUTD); __hip_bfloat16* cgB = bfbuf((size_t)GCn * OUTD);
    float* tg = f32buf((size_t)GTn * OUTD); __hip_bfloat16* tgB = bfbuf((size_t)GTn * OUTD);

    // ---- 1. weight conversion (bf16, transposed, K padded) ----
    cvtw(stream, F(P_CN_W), cnWt, 32, HD, 32);
    cvtw(stream, F(P_TN_W), tnWt, 16, HD, 32);
    for (int l = 0; l < 2; ++l) {
        cvtw(stream, F(P_CG_W) + (size_t)l * 264 * HD, cgWt[l], 264, HD, KGP);
        cvtw(stream, F(P_TG_W) + (size_t)l * 264 * HD, tgWt[l], 264, HD, KGP);
    }
    cvtw(stream, F(P_CF_W), cfWt, HD, OUTD, HD);
    cvtw(stream, F(P_TF_W), tfWt, HD, OUTD, HD);
    const int preBase[2] = { P_CC_BASE, P_TC_BASE };
    for (int p = 0; p < 2; ++p) {
        int b0 = preBase[p];
        for (int l = 0; l < 2; ++l) {
            cvtw(stream, F(b0 + 0)  + (size_t)l * HD * QKD,    wq1t[p][l], HD, QKD, HD);
            cvtw(stream, F(b0 + 2)  + (size_t)l * HD * QKD,    wk1t[p][l], HD, QKD, HD);
            cvtw(stream, F(b0 + 4)  + (size_t)l * HD * HD,     wv1t[p][l], HD, HD, HD);
            cvtw(stream, F(b0 + 6)  + (size_t)l * HD * QKD,    wq2t[p][l], HD, QKD, HD);
            cvtw(stream, F(b0 + 8)  + (size_t)l * HD * QKD,    wk2t[p][l], HD, QKD, HD);
            cvtw(stream, F(b0 + 10) + (size_t)l * HD * HD,     wv2t[p][l], HD, HD, HD);
            cvtw(stream, F(b0 + 12) + (size_t)l * 2 * HD * HD, wdt[p][l],  2 * HD, HD, 2 * HD);
        }
    }

    // ---- 2. node / edge embeddings ----
    k_build_Ac<<<cdiv(NCn * 32, 256), 256, 0, stream>>>(F(IN_CFEATS), F(P_OP_EMB), ctypes, Ac, NCn);
    launch_gemm(stream, Ac, 32, cnWt, 32, F(P_CN_B), NCn, HD, 32,
                cF, HD, nullptr, 0, cB, HD, nullptr, 0, nullptr, 0, 0, nullptr, nullptr, 0, 1);
    k_build_At<<<cdiv(NTn * 32, 256), 256, 0, stream>>>(F(IN_TFEATS), At, NTn);
    launch_gemm(stream, At, 32, tnWt, 32, F(P_TN_B), NTn, HD, 32,
                tF, HD, nullptr, 0, tB, HD, nullptr, 0, nullptr, 0, 0, nullptr, nullptr, 0, 1);
    k_edge_embed<<<cdiv(ECn * 8, 256), 256, 0, stream>>>(F(IN_CEDGE), F(P_CE_W), F(P_CE_B), ceF, ECn);
    k_edge_embed<<<cdiv(ETn * 8, 256), 256, 0, stream>>>(F(IN_TEDGE), F(P_TE_W), F(P_TE_B), teF, ETn);

    // ---- 3. layers ----
    for (int l = 0; l < 2; ++l) {
        // gconv c
        zero_f(stream, aggC, (long long)NCn * KGP);
        k_scatter_edges<<<ECn, 256, 0, stream>>>(cF, HD, ceF, EHD, c_src, c_dst, aggC, KGP, ECn, HD);
        f2b(stream, aggC, aggCB, (long long)NCn * KGP);
        launch_gemm(stream, aggCB, KGP, cgWt[l], KGP, F(P_CG_B) + l * HD, NCn, HD, KGP,
                    cF, HD, cF, HD, cB, HD, nullptr, 0, nullptr, 0, 0, nullptr, nullptr, 0, 1);
        // gconv t
        zero_f(stream, aggT, (long long)NTn * KGP);
        k_scatter_edges<<<ETn, 256, 0, stream>>>(tF, HD, teF, EHD, t_src, t_dst, aggT, KGP, ETn, HD);
        f2b(stream, aggT, aggTB, (long long)NTn * KGP);
        launch_gemm(stream, aggTB, KGP, tgWt[l], KGP, F(P_TG_B) + l * HD, NTn, HD, KGP,
                    tF, HD, tF, HD, tB, HD, nullptr, 0, nullptr, 0, 0, nullptr, nullptr, 0, 1);

        // cross c <- attend over t  (mask = placement[row=c, col=t])
        {
            int b0 = P_CC_BASE;
            __hip_bfloat16* wq[2] = { wq1t[0][l], wq2t[0][l] };
            __hip_bfloat16* wk[2] = { wk1t[0][l], wk2t[0][l] };
            __hip_bfloat16* wv[2] = { wv1t[0][l], wv2t[0][l] };
            const float* bq[2] = { F(b0 + 1) + l * QKD, F(b0 + 7) + l * QKD };
            const float* bk[2] = { F(b0 + 3) + l * QKD, F(b0 + 9) + l * QKD };
            const float* bv[2] = { F(b0 + 5) + l * HD,  F(b0 + 11) + l * HD };
            do_cross(stream, cF, cB, NCn, tB, NTn, wq, bq, wk, bk, wv, bv,
                     wdt[0][l], F(b0 + 13) + l * HD, placement, 1,
                     Qs, Ks, Vt, S, P, cat);
        }
        // cross t <- attend over (updated) c  (mask = placement^T)
        {
            int b0 = P_TC_BASE;
            __hip_bfloat16* wq[2] = { wq1t[1][l], wq2t[1][l] };
            __hip_bfloat16* wk[2] = { wk1t[1][l], wk2t[1][l] };
            __hip_bfloat16* wv[2] = { wv1t[1][l], wv2t[1][l] };
            const float* bq[2] = { F(b0 + 1) + l * QKD, F(b0 + 7) + l * QKD };
            const float* bk[2] = { F(b0 + 3) + l * QKD, F(b0 + 9) + l * QKD };
            const float* bv[2] = { F(b0 + 5) + l * HD,  F(b0 + 11) + l * HD };
            do_cross(stream, tF, tB, NTn, cB, NCn, wq, bq, wk, bk, wv, bv,
                     wdt[1][l], F(b0 + 13) + l * HD, placement, 2,
                     Qs, Ks, Vt, S, P, cat);
        }
    }

    // ---- 4. final projections with fused group segment-sum ----
    zero_f(stream, cg, (long long)GCn * OUTD);
    zero_f(stream, tg, (long long)GTn * OUTD);
    launch_gemm(stream, cB, HD, cfWt, HD, F(P_CF_B), NCn, OUTD, HD,
                nullptr, 0, nullptr, 0, nullptr, 0, nullptr, 0,
                nullptr, 0, 0, c_gid, cg, OUTD, 0);
    launch_gemm(stream, tB, HD, tfWt, HD, F(P_TF_B), NTn, OUTD, HD,
                nullptr, 0, nullptr, 0, nullptr, 0, nullptr, 0,
                nullptr, 0, 0, t_gid, tg, OUTD, 0);
    f2b(stream, cg, cgB, (long long)GCn * OUTD);
    f2b(stream, tg, tgB, (long long)GTn * OUTD);

    // out[GC,GT] = cg @ tg^T  (Bt := tg, already [GT, OUTD])
    launch_gemm(stream, cgB, OUTD, tgB, OUTD, nullptr, GCn, GTn, OUTD,
                (float*)d_out, GTn, nullptr, 0, nullptr, 0, nullptr, 0,
                nullptr, 0, 0, nullptr, nullptr, 0, 0);
}
